// Model_58772332478872
// MI455X (gfx1250) — matrix operand order repaired and checked by reading, not compiled, not run on hardware
//
#include <hip/hip_runtime.h>
#include <hip/hip_bf16.h>

typedef _Float16 h16 __attribute__((ext_vector_type(16)));
typedef _Float16 h8  __attribute__((ext_vector_type(8)));
typedef float    f8  __attribute__((ext_vector_type(8)));

#define HDIM 128
#define ROWS 16            // batch rows per block (one WMMA row-tile)
#define NW   8             // waves per block (wave j owns h-col-tile j)

// ---- fast activations (v_exp_f32 + v_rcp_f32) ----
__device__ __forceinline__ float fast_sigmoid(float x) {
    return __builtin_amdgcn_rcpf(1.0f + __builtin_amdgcn_exp2f(-1.44269504f * x));
}
__device__ __forceinline__ float fast_tanh(float x) {
    return 1.0f - 2.0f * __builtin_amdgcn_rcpf(1.0f + __builtin_amdgcn_exp2f(2.88539008f * x));
}

// A fragment 16x32 f16, row-major src (ld elems). ISA layout:
// lane<16: K 0..7 (v0-3) + 16..23 (v4-7); lane>=16: K 8..15 + 24..31.
__device__ __forceinline__ h16 load_frag_a(const _Float16* p, int ld, int k0, int lane) {
    const int r = lane & 15, ks = lane >> 4;
    const _Float16* base = p + r * ld + k0 + 8 * ks;
    h8 lo = *(const h8*)(base);
    h8 hi = *(const h8*)(base + 16);
    return __builtin_shufflevector(lo, hi, 0,1,2,3,4,5,6,7,8,9,10,11,12,13,14,15);
}

// B fragment 32x16 f16 from row-major W[N][K] (B(k,n)=W[n][k]).
// lane<16: col n0+lane, K k0..k0+7 + k0+16..k0+23; lane>=16: K k0+8..k0+15 + k0+24..k0+31.
__device__ __forceinline__ h16 load_frag_b(const _Float16* p, int ld, int n0, int k0, int lane) {
    const int r = lane & 15, ks = lane >> 4;
    const _Float16* base = p + (n0 + r) * ld + k0 + 8 * ks;
    h8 lo = *(const h8*)(base);
    h8 hi = *(const h8*)(base + 16);
    return __builtin_shufflevector(lo, hi, 0,1,2,3,4,5,6,7,8,9,10,11,12,13,14,15);
}

#define WMMA(a, b, c) __builtin_amdgcn_wmma_f32_16x16x32_f16(false, (a), false, (b), (short)0, (c), false, false)

// ---- pre-pass: convert the three [4H,H] fp32 weights to f16 in workspace ----
__global__ void cvt_w_f16(const float* __restrict__ w1, const float* __restrict__ w2,
                          const float* __restrict__ w3, _Float16* __restrict__ dst) {
    int i = blockIdx.x * 256 + threadIdx.x;
    const int N = 4 * HDIM * HDIM;  // 65536
    if (i < N)              dst[i] = (_Float16)w1[i];
    else if (i < 2 * N)     dst[i] = (_Float16)w2[i - N];
    else if (i < 3 * N)     dst[i] = (_Float16)w3[i - 2 * N];
}

__global__ __launch_bounds__(256, 1) void lstm2_persistent(
    const float* __restrict__ X,
    const float* __restrict__ Wih1,   // [4H,1] flat [512]
    const float* __restrict__ bih1, const float* __restrict__ bhh1,
    const float* __restrict__ bih2, const float* __restrict__ bhh2,
    const float* __restrict__ Wlin,   // [1,H] flat [128]
    const float* __restrict__ blin,
    const _Float16* __restrict__ Whh1h,   // [4H,H] f16 row-major
    const _Float16* __restrict__ Wih2h,
    const _Float16* __restrict__ Whh2h,
    float* __restrict__ out, int T)
{
    __shared__ __align__(16) _Float16 h1buf[ROWS * HDIM];
    __shared__ __align__(16) _Float16 h2buf[ROWS * HDIM];
    __shared__ __align__(16) float    xbuf[ROWS];
    __shared__ __align__(16) float    wlin_s[HDIM];
    __shared__ __align__(16) float    predpart[16][ROWS];

    const int tid  = threadIdx.x;
    const int lane = tid & 31;
    const int wave = tid >> 5;       // h col-tile index j = 0..7
    const int rl   = lane & 15;
    const int ks   = lane >> 4;
    const int row0 = blockIdx.x * ROWS;

    for (int i = tid; i < ROWS * HDIM; i += 256) {
        h1buf[i] = (_Float16)0.0f;
        h2buf[i] = (_Float16)0.0f;
    }
    if (tid < HDIM) wlin_s[tid] = Wlin[tid];

    // per-lane constants: combined biases + layer-1 input weight column
    float b1s[4], b2s[4], wx1[4];
#pragma unroll
    for (int q = 0; q < 4; ++q) {
        const int col = q * HDIM + wave * 16 + rl;
        b1s[q] = bih1[col] + bhh1[col];
        b2s[q] = bih2[col] + bhh2[col];
        wx1[q] = Wih1[col];
    }
    const float blin_v = blin[0];

    // cell state, resident in registers in C-matrix layout (row = v + 8*ks)
    f8 c1, c2;
#pragma unroll
    for (int v = 0; v < 8; ++v) { c1[v] = 0.0f; c2[v] = 0.0f; }

    __syncthreads();

    for (int t = 0; t < T; ++t) {
        // ---- stage x_t ----
        if (tid < ROWS) xbuf[tid] = X[(row0 + tid) * T + t];
        __syncthreads();

        // ---- layer 1: gates = b1 + x*Wih1 + h1 @ Whh1^T ----
        f8 acc[4];
        float xv[8];
        {
            const float4* xp = (const float4*)&xbuf[8 * ks];
            float4 xa = xp[0], xb = xp[1];
            xv[0] = xa.x; xv[1] = xa.y; xv[2] = xa.z; xv[3] = xa.w;
            xv[4] = xb.x; xv[5] = xb.y; xv[6] = xb.z; xv[7] = xb.w;
        }
#pragma unroll
        for (int q = 0; q < 4; ++q)
#pragma unroll
            for (int v = 0; v < 8; ++v)
                acc[q][v] = b1s[q] + xv[v] * wx1[q];

#pragma unroll
        for (int k = 0; k < HDIM; k += 32) {
            h16 a = load_frag_a(h1buf, HDIM, k, lane);
#pragma unroll
            for (int q = 0; q < 4; ++q) {
                h16 b = load_frag_b(Whh1h, HDIM, q * HDIM + wave * 16, k, lane);
                acc[q] = WMMA(a, b, acc[q]);
            }
        }
        __syncthreads();   // all reads of h1buf (t-1) complete

        // pointwise LSTM cell 1; write h1 (f16) to LDS
#pragma unroll
        for (int v = 0; v < 8; ++v) {
            float ig = fast_sigmoid(acc[0][v]);
            float fg = fast_sigmoid(acc[1][v]);
            float gg = fast_tanh(acc[2][v]);
            float og = fast_sigmoid(acc[3][v]);
            float c  = fg * c1[v] + ig * gg;
            c1[v] = c;
            float h  = og * fast_tanh(c);
            h1buf[(v + 8 * ks) * HDIM + wave * 16 + rl] = (_Float16)h;
        }
        __syncthreads();

        // ---- layer 2: gates = b2 + h1 @ Wih2^T + h2 @ Whh2^T ----
#pragma unroll
        for (int q = 0; q < 4; ++q)
#pragma unroll
            for (int v = 0; v < 8; ++v)
                acc[q][v] = b2s[q];

#pragma unroll
        for (int k = 0; k < HDIM; k += 32) {
            h16 a = load_frag_a(h1buf, HDIM, k, lane);
#pragma unroll
            for (int q = 0; q < 4; ++q) {
                h16 b = load_frag_b(Wih2h, HDIM, q * HDIM + wave * 16, k, lane);
                acc[q] = WMMA(a, b, acc[q]);
            }
        }
#pragma unroll
        for (int k = 0; k < HDIM; k += 32) {
            h16 a = load_frag_a(h2buf, HDIM, k, lane);
#pragma unroll
            for (int q = 0; q < 4; ++q) {
                h16 b = load_frag_b(Whh2h, HDIM, q * HDIM + wave * 16, k, lane);
                acc[q] = WMMA(a, b, acc[q]);
            }
        }
        __syncthreads();   // all reads of h2buf (t-1) complete

        // pointwise LSTM cell 2; write h2 (f16) to LDS
#pragma unroll
        for (int v = 0; v < 8; ++v) {
            float ig = fast_sigmoid(acc[0][v]);
            float fg = fast_sigmoid(acc[1][v]);
            float gg = fast_tanh(acc[2][v]);
            float og = fast_sigmoid(acc[3][v]);
            float c  = fg * c2[v] + ig * gg;
            c2[v] = c;
            float h  = og * fast_tanh(c);
            h2buf[(v + 8 * ks) * HDIM + wave * 16 + rl] = (_Float16)h;
        }
        __syncthreads();

        // ---- linear head: pred[b] = h2[b,:] . Wlin + blin (16-way split) ----
        {
            const int b   = tid & (ROWS - 1);
            const int seg = tid >> 4;       // 0..15 -> 8 h each
            float s = 0.0f;
#pragma unroll
            for (int h = 0; h < 8; ++h) {
                const int hh = seg * 8 + h;
                s += (float)h2buf[b * HDIM + hh] * wlin_s[hh];
            }
            predpart[seg][b] = s;
        }
        __syncthreads();
        if (tid < ROWS) {
            float p = blin_v;
#pragma unroll
            for (int q = 0; q < 16; ++q) p += predpart[q][tid];
            out[(row0 + tid) * T + t] = p;
        }
        // next write to predpart is behind >=3 barriers of step t+1; safe
    }
}

extern "C" void kernel_launch(void* const* d_in, const int* in_sizes, int n_in,
                              void* d_out, int out_size, void* d_ws, size_t ws_size,
                              hipStream_t stream) {
    const float* X    = (const float*)d_in[0];
    const float* Wih1 = (const float*)d_in[1];
    const float* Whh1 = (const float*)d_in[2];
    const float* bih1 = (const float*)d_in[3];
    const float* bhh1 = (const float*)d_in[4];
    const float* Wih2 = (const float*)d_in[5];
    const float* Whh2 = (const float*)d_in[6];
    const float* bih2 = (const float*)d_in[7];
    const float* bhh2 = (const float*)d_in[8];
    const float* Wlin = (const float*)d_in[9];
    const float* blin = (const float*)d_in[10];

    const int B = 512;
    const int T = in_sizes[0] / B;
    const int N = 4 * HDIM * HDIM;  // 65536 elems per weight matrix

    _Float16* wsh = (_Float16*)d_ws;   // [3*N] f16 weights

    cvt_w_f16<<<(3 * N + 255) / 256, 256, 0, stream>>>(Whh1, Wih2, Whh2, wsh);

    lstm2_persistent<<<B / ROWS, 256, 0, stream>>>(
        X, Wih1, bih1, bhh1, bih2, bhh2, Wlin, blin,
        wsh, wsh + N, wsh + 2 * N, (float*)d_out, T);
}
